// GraphMAE_28114855919780
// MI455X (gfx1250) — compile-verified, matrix operand order checked by reading, not run on hardware
//
#include <hip/hip_runtime.h>
#include <hip/hip_bf16.h>
#include <stdint.h>

// ---------------------------------------------------------------------------
// GraphMAE on MI455X (gfx1250, wave32).
// GEMMs: v_wmma_f32_16x16x32_bf16, one wave per 16x16 tile, K=128 (4 WMMAs).
// Weights are pre-swizzled once into WMMA B-fragment layout so every operand
// load in the GEMM inner loop is a 128-bit global load. Epilogue variants are
// template-specialized so bias/ReLU/store paths are branch-free.
// Edge aggregation: float4 gathers + fp32 global atomics. All accumulation
// fp32; only WMMA A/B operands are bf16.
// ---------------------------------------------------------------------------

typedef __attribute__((ext_vector_type(16))) __bf16 v16bf;
typedef __attribute__((ext_vector_type(8)))  float  v8f;

struct bf4 { __bf16 x, y, z, w; };   // 8-byte packed bf16 quad

#define D 128
#define THREADS 256

// ---------------- utility kernels ----------------

__global__ void k_zero(float* __restrict__ p, int n) {
    int i = blockIdx.x * blockDim.x + threadIdx.x;
    if (i < n) p[i] = 0.0f;
}

__global__ void k_f2b(const float* __restrict__ in, __bf16* __restrict__ out, int n) {
    int i = (blockIdx.x * blockDim.x + threadIdx.x) * 4;
    if (i + 3 < n) {
        float4 v = *(const float4*)(in + i);
        bf4 o = { (__bf16)v.x, (__bf16)v.y, (__bf16)v.z, (__bf16)v.w };
        *(bf4*)(out + i) = o;
    } else {
        for (; i < n; ++i) out[i] = (__bf16)in[i];
    }
}

// ---------------- degree / normalization ----------------

__global__ void k_degree(const long long* __restrict__ dst, float* __restrict__ deg, int E) {
    int e = blockIdx.x * blockDim.x + threadIdx.x;
    if (e < E) atomicAdd(&deg[dst[e]], 1.0f);
}

// in-place: deg -> rsqrt(deg + 1)
__global__ void k_dinv(float* __restrict__ deg, int n) {
    int i = blockIdx.x * blockDim.x + threadIdx.x;
    if (i < n) deg[i] = rsqrtf(deg[i] + 1.0f);
}

// ---------------- weight pre-swizzle into WMMA B-fragment layout ----------------
// Fragment buffer layout: [tile 0..7][kb 0..3][lane 0..31][16 bf16] = 32 KB.
// Per (tile,kb,lane): lane<16 holds col=tile*16+lane, K{kb*32+0..7, kb*32+16..23};
// lane>=16 holds col=tile*16+lane-16, K{kb*32+8..15, kb*32+24..31}.
__global__ void k_prep_w(const float* __restrict__ W, __bf16* __restrict__ F) {
    const int t = blockIdx.x * blockDim.x + threadIdx.x;
    if (t >= 8 * 4 * 32) return;
    const int lane = t & 31;
    const int kb   = (t >> 5) & 3;
    const int tile = t >> 7;
    const int half = lane >> 4, l16 = lane & 15;
    const int col  = tile * 16 + l16;
    const int ka0  = kb * 32 + (half ? 8 : 0);
    const int ka1  = kb * 32 + (half ? 24 : 16);
    union { bf4 q[4]; __bf16 e[16]; } o;
    #pragma unroll
    for (int i = 0; i < 8; ++i) o.e[i]     = (__bf16)W[(ka0 + i) * D + col];
    #pragma unroll
    for (int i = 0; i < 8; ++i) o.e[8 + i] = (__bf16)W[(ka1 + i) * D + col];
    bf4* dstp = (bf4*)(F + (size_t)t * 16);
    #pragma unroll
    for (int i = 0; i < 4; ++i) dstp[i] = o.q[i];
}

// ---------------- WMMA GEMM: [M,128] x [128,128] (+bias)(+relu) ----------------
// Block = 256 threads = 8 waves; block handles 16 rows, wave w -> cols [16w,16w+16).
// Bf is the pre-swizzled fragment buffer (k_prep_w). All operand loads are b128.
// Template flags keep the epilogue branch-free.
template <int HAS_BIAS, int RELU, int WF, int WB>
__global__ __launch_bounds__(THREADS) void k_gemm128(
    const __bf16* __restrict__ A,     // [M,128] row-major bf16
    const __bf16* __restrict__ Bf,    // fragment-layout weights (32 KB)
    const float*  __restrict__ bias,  // [128] (used iff HAS_BIAS)
    float*        __restrict__ outF,  // [M,128] (used iff WF)
    __bf16*       __restrict__ outB)  // [M,128] (used iff WB)
{
    const int wave = threadIdx.x >> 5;
    const int lane = threadIdx.x & 31;
    const int half = lane >> 4;            // 0: lanes 0-15, 1: lanes 16-31
    const int l16  = lane & 15;
    const int rowBase = blockIdx.x << 4;   // M must be a multiple of 16 (100000 is)
    const int col     = (wave << 4) + l16;

    const __bf16* arow = A + (size_t)(rowBase + l16) * D;
    const __bf16* bf   = Bf + ((size_t)(wave * 4) * 32 + lane) * 16;

    v8f acc = {};
    #pragma unroll
    for (int kb = 0; kb < 4; ++kb) {
        // ISA A layout (16-bit 16x32): lane<16 holds K{kb*32+0..7, +16..23},
        // lane>=16 holds K{kb*32+8..15, +24..31}; each chunk contiguous -> b128.
        const int ka0 = kb * 32 + (half ? 8 : 0);
        const int ka1 = kb * 32 + (half ? 24 : 16);
        union { v16bf v; uint4 q[2]; } a, b;
        a.q[0] = *(const uint4*)(arow + ka0);
        a.q[1] = *(const uint4*)(arow + ka1);
        const __bf16* bp = bf + (size_t)kb * 32 * 16;
        b.q[0] = *(const uint4*)(bp);
        b.q[1] = *(const uint4*)(bp + 8);
        acc = __builtin_amdgcn_wmma_f32_16x16x32_bf16(
            /*neg_a=*/false, a.v, /*neg_b=*/false, b.v,
            /*c_mod=*/(short)0, acc, /*reuse_a=*/false, /*reuse_b=*/false);
    }

    const float bval = HAS_BIAS ? bias[col] : 0.0f;
    // C/D layout: lane<16 -> rows rowBase+0..7 (N=l16), lane>=16 -> rows rowBase+8..15.
    #pragma unroll
    for (int r = 0; r < 8; ++r) {
        const int row = rowBase + (half << 3) + r;
        float v = HAS_BIAS ? (acc[r] + bval) : acc[r];   // no dead +0.0f in no-bias variant
        if (RELU) v = fmaxf(v, 0.0f);
        const size_t idx = (size_t)row * D + col;
        if (WF) outF[idx] = v;
        if (WB) outB[idx] = (__bf16)v;
    }
}

// ---------------- GCN aggregate ----------------

// agg[v,:] = h[v,:] * dinv[v]^2 + bias[:]
__global__ void k_aggr_init(const float* __restrict__ h, const float* __restrict__ dinv,
                            const float* __restrict__ bias, float* __restrict__ agg, int total) {
    int i = (blockIdx.x * blockDim.x + threadIdx.x) * 4;
    if (i >= total) return;
    const int row = i >> 7, c = i & (D - 1);
    float w = dinv[row]; w *= w;
    float4 hv = *(const float4*)(h + i);
    float4 bv = *(const float4*)(bias + c);
    float4 o  = { hv.x * w + bv.x, hv.y * w + bv.y, hv.z * w + bv.z, hv.w * w + bv.w };
    *(float4*)(agg + i) = o;
}

// one wave per edge: agg[dst,:] += h[src,:] * dinv[src]*dinv[dst]
__global__ void k_scatter(const long long* __restrict__ src, const long long* __restrict__ dst,
                          const float* __restrict__ dinv, const float* __restrict__ h,
                          float* __restrict__ agg, int E) {
    const int e = blockIdx.x * (blockDim.x >> 5) + (threadIdx.x >> 5);
    if (e >= E) return;
    const int lane = threadIdx.x & 31;
    const long long s = src[e];
    const long long d = dst[e];
    const float norm = dinv[s] * dinv[d];
    float4 v = *(const float4*)(h + (size_t)s * D + lane * 4);
    float* ad = agg + (size_t)d * D + lane * 4;
    atomicAdd(ad + 0, v.x * norm);
    atomicAdd(ad + 1, v.y * norm);
    atomicAdd(ad + 2, v.z * norm);
    atomicAdd(ad + 3, v.w * norm);
}

// relu in fp32; emit bf16 copy for next GEMM, optional unaligned-safe fp32 copy
template <int WF>
__global__ void k_relu_cvt(const float* __restrict__ in, float* __restrict__ outF,
                           __bf16* __restrict__ outB, int total) {
    int i = (blockIdx.x * blockDim.x + threadIdx.x) * 4;
    if (i >= total) return;
    float4 v = *(const float4*)(in + i);
    v.x = fmaxf(v.x, 0.0f); v.y = fmaxf(v.y, 0.0f);
    v.z = fmaxf(v.z, 0.0f); v.w = fmaxf(v.w, 0.0f);
    if (WF) {  // d_out+1 is only 4B aligned -> scalar stores
        outF[i + 0] = v.x; outF[i + 1] = v.y; outF[i + 2] = v.z; outF[i + 3] = v.w;
    }
    bf4 o = { (__bf16)v.x, (__bf16)v.y, (__bf16)v.z, (__bf16)v.w };
    *(bf4*)(outB + i) = o;
}

// ---------------- loss ----------------

__global__ void k_mask_count(const unsigned char* __restrict__ mask, float* __restrict__ acc, int n) {
    int i = blockIdx.x * blockDim.x + threadIdx.x;
    float v = (i < n && mask[i]) ? 1.0f : 0.0f;
    #pragma unroll
    for (int off = 16; off > 0; off >>= 1) v += __shfl_down(v, off);
    if ((threadIdx.x & 31) == 0 && v != 0.0f) atomicAdd(acc + 1, v);
}

__global__ void k_loss(const float* __restrict__ r, const float* __restrict__ x,
                       const unsigned char* __restrict__ mask, float* __restrict__ acc,
                       long long total) {
    const long long stride = (long long)gridDim.x * blockDim.x;
    float v = 0.0f;
    for (long long idx = (long long)blockIdx.x * blockDim.x + threadIdx.x;
         idx < total; idx += stride) {
        const int row = (int)(idx >> 7);
        if (mask[row]) { float dlt = r[idx] - x[idx]; v += dlt * dlt; }
    }
    #pragma unroll
    for (int off = 16; off > 0; off >>= 1) v += __shfl_down(v, off);
    __shared__ float red[8];
    const int wid = threadIdx.x >> 5, lane = threadIdx.x & 31;
    if (lane == 0) red[wid] = v;
    __syncthreads();
    if (wid == 0) {
        float t = (lane < 8) ? red[lane] : 0.0f;
        #pragma unroll
        for (int off = 4; off > 0; off >>= 1) t += __shfl_down(t, off);
        if (lane == 0) atomicAdd(acc, t);
    }
}

__global__ void k_final(const float* __restrict__ acc, float* __restrict__ out) {
    out[0] = acc[0] / (fmaxf(acc[1], 1.0f) * (float)D);
}

// ---------------- launcher ----------------

extern "C" void kernel_launch(void* const* d_in, const int* in_sizes, int n_in,
                              void* d_out, int out_size, void* d_ws, size_t ws_size,
                              hipStream_t stream) {
    // inputs: x, edge_index, mask, W1,b1, W2,b2, W3,b3, D1,db1, D2,db2, D3,db3
    const float*         x    = (const float*)d_in[0];
    const long long*     ei   = (const long long*)d_in[1];
    const unsigned char* mask = (const unsigned char*)d_in[2];  // jax bool = 1 byte
    const float* W[3]  = { (const float*)d_in[3], (const float*)d_in[5], (const float*)d_in[7] };
    const float* bc[3] = { (const float*)d_in[4], (const float*)d_in[6], (const float*)d_in[8] };
    const float* Dw[3] = { (const float*)d_in[9], (const float*)d_in[11], (const float*)d_in[13] };
    const float* db[3] = { (const float*)d_in[10], (const float*)d_in[12], (const float*)d_in[14] };

    const int Nn = in_sizes[0] / D;       // 100000 (multiple of 16)
    const int E  = in_sizes[1] / 2;       // 1600000
    const long long total = (long long)Nn * D;

    const long long* src = ei;
    const long long* dst = ei + E;

    // workspace carve-up
    char* base = (char*)d_ws;
    size_t off = 0;
    auto take = [&](size_t bytes) -> void* {
        void* p = base + off;
        off = (off + bytes + 255) & ~(size_t)255;
        return p;
    };
    float*  dinv = (float*)take((size_t)Nn * 4);            //  0.4 MB (deg, then rsqrt in place)
    float*  h    = (float*)take((size_t)Nn * D * 4);        // 51.2 MB
    float*  agg  = (float*)take((size_t)Nn * D * 4);        // 51.2 MB
    __bf16* xb   = (__bf16*)take((size_t)Nn * D * 2);       // 25.6 MB
    __bf16* wb   = (__bf16*)take((size_t)6 * D * D * 2);    //  0.2 MB (6 frag-layout mats)
    float*  acc  = (float*)take(256);                       // [0]=sq sum, [1]=mask count
    __bf16* hb   = (__bf16*)h;                              // decoder bf16 scratch aliases h

    float* zout = (float*)d_out + 1;                        // z goes after the scalar loss

    const int gN    = (Nn + THREADS - 1) / THREADS;         // per-node grids
    const int gE    = (E + THREADS - 1) / THREADS;
    const int gVec  = (int)((total / 4 + THREADS - 1) / THREADS);  // per-4-element grids
    const int gGemm = Nn / 16;                              // one block per 16 rows
    const int gScat = (E + 7) / 8;                          // one wave per edge
    const int gPrep = (8 * 4 * 32 + THREADS - 1) / THREADS; // fragment swizzle

    // degrees + symmetric normalization
    k_zero<<<gN, THREADS, 0, stream>>>(dinv, Nn);
    k_zero<<<1, 64, 0, stream>>>(acc, 64);
    k_degree<<<gE, THREADS, 0, stream>>>(dst, dinv, E);
    k_dinv<<<gN, THREADS, 0, stream>>>(dinv, Nn);

    // operand prep: activations -> bf16 rows, weights -> bf16 WMMA fragments
    k_f2b<<<gVec, THREADS, 0, stream>>>(x, xb, (int)total);
    for (int l = 0; l < 3; ++l) {
        k_prep_w<<<gPrep, THREADS, 0, stream>>>(W[l],  wb + (size_t)l * D * D);
        k_prep_w<<<gPrep, THREADS, 0, stream>>>(Dw[l], wb + (size_t)(3 + l) * D * D);
    }

    // encoder: 3x (GCNConv -> ReLU)
    for (int l = 0; l < 3; ++l) {
        k_gemm128<0, 0, 1, 0><<<gGemm, THREADS, 0, stream>>>(
            xb, wb + (size_t)l * D * D, nullptr, h, nullptr);
        k_aggr_init<<<gVec, THREADS, 0, stream>>>(h, dinv, bc[l], agg, (int)total);
        k_scatter<<<gScat, THREADS, 0, stream>>>(src, dst, dinv, h, agg, E);
        if (l == 2)
            k_relu_cvt<1><<<gVec, THREADS, 0, stream>>>(agg, zout, xb, (int)total);
        else
            k_relu_cvt<0><<<gVec, THREADS, 0, stream>>>(agg, nullptr, xb, (int)total);
    }

    // decoder MLP (bias+ReLU fused into WMMA epilogue)
    k_gemm128<1, 1, 0, 1><<<gGemm, THREADS, 0, stream>>>(
        xb, wb + (size_t)3 * D * D, db[0], nullptr, hb);
    k_gemm128<1, 1, 0, 1><<<gGemm, THREADS, 0, stream>>>(
        hb, wb + (size_t)4 * D * D, db[1], nullptr, xb);
    k_gemm128<1, 0, 1, 0><<<gGemm, THREADS, 0, stream>>>(
        xb, wb + (size_t)5 * D * D, db[2], agg, nullptr);

    // masked MSE
    k_mask_count<<<gN, THREADS, 0, stream>>>(mask, acc, Nn);
    k_loss<<<2048, THREADS, 0, stream>>>(agg, x, mask, acc, total);
    k_final<<<1, 1, 0, stream>>>(acc, (float*)d_out);
}
